// SRToneMapper_11931419148447
// MI455X (gfx1250) — matrix operand order repaired and checked by reading, not compiled, not run on hardware
//
#include <hip/hip_runtime.h>

// ---------------- MI455X / gfx1250: wave32, WMMA f32_16x16x32_f16 ----------------

typedef __attribute__((ext_vector_type(16))) _Float16 v16h;
typedef __attribute__((ext_vector_type(8)))  _Float16 v8h;
typedef __attribute__((ext_vector_type(8)))  float    v8f;

#define HDIM 128
#define WDIM 256
#define HWPIX (HDIM * WDIM)   // 32768
#define QDIM  32768

// A fragment (16x32 f16, M = lane&15): lanes 0-15 hold K 0..15, lanes 16-31 hold K 16..31
// -> one contiguous 32B load.
__device__ __forceinline__ v16h load_a_frag(const _Float16* xrow, int ks, int lane) {
  return *(const v16h*)(xrow + ks * 32 + ((lane >> 4) << 4));
}
// B fragment (32x16 f16, N = lane&15): lanes 0-15 hold K 0..15, lanes 16-31 hold K 16..31
// -> one contiguous 32B load from the pre-transposed weight row Wt[n][k].
__device__ __forceinline__ v16h load_b_frag(const _Float16* wrow, int ks, int lane) {
  return *(const v16h*)(wrow + ks * 32 + ((lane >> 4) << 4));
}
__device__ __forceinline__ v8f wmma_step(v16h a, v16h b, v8f c) {
  return __builtin_amdgcn_wmma_f32_16x16x32_f16(false, a, false, b, (short)0, c, false, false);
}

// Generic [16,K] x Wt[64*ngroups,K]^T -> LDS [16,os] f16 (bias + optional ReLU).
// N is processed in groups of 4 x 16 so one A fragment feeds 4 WMMAs per K-step;
// K-loop unroll is capped so A-frag LDS loads are not hoisted+spilled to scratch.
__device__ __forceinline__ void gemm_lds(const _Float16* __restrict__ xt, int xs,
                                         const _Float16* __restrict__ wt, int ws,
                                         const float* __restrict__ bias, int nout,
                                         int ksteps, int ngroups,
                                         _Float16* __restrict__ out, int os,
                                         bool relu, int lane) {
  const int m = lane & 15;
  const int hi = lane >> 4;
  const _Float16* xrow = xt + (size_t)m * xs;
  for (int ng = 0; ng < ngroups; ++ng) {
    v8f acc[4];
#pragma unroll
    for (int t = 0; t < 4; ++t) {
      int n = (ng * 4 + t) * 16 + m;
      float bv = (n < nout) ? bias[n] : 0.0f;
      acc[t] = (v8f){bv, bv, bv, bv, bv, bv, bv, bv};
    }
    const _Float16* wrow = wt + (size_t)((ng * 4) * 16 + m) * ws;
    __builtin_prefetch(wrow, 0, 0);
#pragma clang loop unroll_count(2)
    for (int ks = 0; ks < ksteps; ++ks) {
      v16h a = load_a_frag(xrow, ks, lane);
#pragma unroll
      for (int t = 0; t < 4; ++t) {
        v16h b = load_b_frag(wrow + (size_t)t * 16 * ws, ks, lane);
        acc[t] = wmma_step(a, b, acc[t]);
      }
    }
#pragma unroll
    for (int t = 0; t < 4; ++t) {
      int n = (ng * 4 + t) * 16 + m;
#pragma unroll
      for (int v = 0; v < 8; ++v) {
        int row = v + hi * 8;
        float f = acc[t][v];
        if (relu) f = fmaxf(f, 0.0f);
        out[(size_t)row * os + n] = (_Float16)f;
      }
    }
  }
}

// ---------------- encoder: 3x3 conv as implicit GEMM ----------------
// in: [HW][CIN] f16, wt: [64][KPAD] f16 (k = tap*CIN + ci), out: [HW][64] f16
// MODE: 0 plain, 1 relu, 2 out += conv, 3 out = conv + addsrc
template <int CIN, int MODE>
__global__ __launch_bounds__(64, 1) void conv3x3_k(const _Float16* __restrict__ in,
                                                   const _Float16* __restrict__ wt,
                                                   const float* __restrict__ bias,
                                                   _Float16* __restrict__ out,
                                                   const _Float16* __restrict__ addsrc) {
  constexpr int KPAD = ((CIN * 9 + 31) / 32) * 32;
  constexpr int KSTEPS = KPAD / 32;
  __shared__ alignas(32) _Float16 atile[2][16][KPAD];
  const int lane = threadIdx.x & 31;
  const int wv = threadIdx.x >> 5;
  const int ptile = blockIdx.x * 32 + wv * 16;

  if constexpr (CIN == 64) {
    for (int e = lane; e < 16 * 9 * 8; e += 32) {   // 8-half vector chunks
      int mm = e / 72; int r = e - mm * 72;
      int tap = r >> 3; int cg = (r & 7) << 3;
      int p = ptile + mm;
      int y = p >> 8; int x = p & (WDIM - 1);
      int ny = y + tap / 3 - 1, nx = x + (tap % 3) - 1;
      v8h val = {(_Float16)0, (_Float16)0, (_Float16)0, (_Float16)0,
                 (_Float16)0, (_Float16)0, (_Float16)0, (_Float16)0};
      if (ny >= 0 && ny < HDIM && nx >= 0 && nx < WDIM)
        val = *(const v8h*)(in + (size_t)(ny * WDIM + nx) * CIN + cg);
      *(v8h*)(&atile[wv][mm][tap * CIN + cg]) = val;
    }
  } else {
    for (int e = lane; e < 16 * KPAD; e += 32) {
      int mm = e / KPAD; int k = e - mm * KPAD;
      _Float16 v = (_Float16)0;
      if (k < 9 * CIN) {
        int tap = k / CIN, ci = k - tap * CIN;
        int p = ptile + mm;
        int y = p >> 8; int x = p & (WDIM - 1);
        int ny = y + tap / 3 - 1, nx = x + (tap % 3) - 1;
        if (ny >= 0 && ny < HDIM && nx >= 0 && nx < WDIM)
          v = in[(size_t)(ny * WDIM + nx) * CIN + ci];
      }
      atile[wv][mm][k] = v;
    }
  }
  __syncthreads();

  const int m = lane & 15, hi = lane >> 4;
  v8f acc[4];
#pragma unroll
  for (int nt = 0; nt < 4; ++nt) {
    float bv = bias[nt * 16 + m];
    acc[nt] = (v8f){bv, bv, bv, bv, bv, bv, bv, bv};
  }
#pragma clang loop unroll_count(2)
  for (int ks = 0; ks < KSTEPS; ++ks) {
    v16h a = load_a_frag(&atile[wv][m][0], ks, lane);
#pragma unroll
    for (int nt = 0; nt < 4; ++nt) {
      v16h b = load_b_frag(wt + (size_t)(nt * 16 + m) * KPAD, ks, lane);
      acc[nt] = wmma_step(a, b, acc[nt]);
    }
  }
#pragma unroll
  for (int nt = 0; nt < 4; ++nt) {
    int n = nt * 16 + m;
#pragma unroll
    for (int v = 0; v < 8; ++v) {
      int row = v + hi * 8;
      size_t oidx = (size_t)(ptile + row) * 64 + n;
      float f = acc[nt][v];
      if (MODE == 1) f = fmaxf(f, 0.0f);
      if (MODE == 2) f += (float)out[oidx];
      if (MODE == 3) f += (float)addsrc[oidx];
      out[oidx] = (_Float16)f;
    }
  }
}

// ---------------- LIIF query MLP (4 corners fused) ----------------
__global__ __launch_bounds__(32, 1) void sr_query_k(
    const _Float16* __restrict__ feat,  // [HW][64]
    const float* __restrict__ local,    // [Q][2]
    const float* __restrict__ cell,     // [Q][2]
    const _Float16* __restrict__ w0, const float* __restrict__ b0,
    const _Float16* __restrict__ w1, const float* __restrict__ b1,
    const _Float16* __restrict__ w2, const float* __restrict__ b2,
    const _Float16* __restrict__ w3, const float* __restrict__ b3,
    const _Float16* __restrict__ w4, const float* __restrict__ b4,
    float* __restrict__ out_ldr,        // [Q][3]
    _Float16* __restrict__ f4h) {       // [Q][256]
  __shared__ alignas(32) _Float16 xt[16][608];
  __shared__ alignas(32) _Float16 hb[16][256];
  __shared__ alignas(32) _Float16 hc[16][256];
  __shared__ float accI[16][256];
  __shared__ float accP[16][4];
  __shared__ int   s_iy[4][16], s_ix[4][16];
  __shared__ float s_r0[4][16], s_r1[4][16], s_w[4][16];
  __shared__ float s_rc0[16], s_rc1[16];

  const int lane = threadIdx.x;
  const int qbase = blockIdx.x * 16;

  if (lane < 16) {
    int q = qbase + lane;
    float ly = local[q * 2 + 0], lx = local[q * 2 + 1];
    const float eps = 1e-6f;
    float areas[4];
    for (int c = 0; c < 4; ++c) {
      float vx = (c < 2) ? -1.f : 1.f;
      float vy = (c & 1) ? 1.f : -1.f;
      float gy = ly + vx * (1.0f / HDIM) + eps;
      float gx = lx + vy * (1.0f / WDIM) + eps;
      gy = fminf(fmaxf(gy, -1.f + eps), 1.f - eps);
      gx = fminf(fmaxf(gx, -1.f + eps), 1.f - eps);
      int iy = (int)floorf(((gy + 1.f) * HDIM - 1.f) * 0.5f + 0.5f);
      int ix = (int)floorf(((gx + 1.f) * WDIM - 1.f) * 0.5f + 0.5f);
      iy = iy < 0 ? 0 : (iy > HDIM - 1 ? HDIM - 1 : iy);
      ix = ix < 0 ? 0 : (ix > WDIM - 1 ? WDIM - 1 : ix);
      float qy = -1.f + (2.f * iy + 1.f) / HDIM;
      float qx = -1.f + (2.f * ix + 1.f) / WDIM;
      float r0 = (ly - qy) * HDIM;
      float r1 = (lx - qx) * WDIM;
      s_iy[c][lane] = iy; s_ix[c][lane] = ix;
      s_r0[c][lane] = r0; s_r1[c][lane] = r1;
      areas[c] = fabsf(r0 * r1) + 1e-9f;
    }
    float tot = areas[0] + areas[1] + areas[2] + areas[3];
    for (int c = 0; c < 4; ++c) s_w[c][lane] = areas[3 - c] / tot;  // opposite-corner weights
    s_rc0[lane] = cell[q * 2 + 0] * HDIM;
    s_rc1[lane] = cell[q * 2 + 1] * WDIM;
    for (int j = 0; j < 4; ++j) accP[lane][j] = 0.f;
  }
  for (int e = lane; e < 16 * 256; e += 32) ((float*)accI)[e] = 0.f;
  __syncthreads();

  for (int c = 0; c < 4; ++c) {
    // gather input: k = ch*9 + tap (unfold order), then rel(2), rel_cell(2), pad->608
    for (int e = lane; e < 16 * 608; e += 32) {
      int mm = e / 608; int k = e - mm * 608;
      _Float16 v = (_Float16)0;
      if (k < 576) {
        int ch = k / 9, tap = k - ch * 9;
        int ny = s_iy[c][mm] + tap / 3 - 1;
        int nx = s_ix[c][mm] + (tap % 3) - 1;
        if (ny >= 0 && ny < HDIM && nx >= 0 && nx < WDIM)
          v = feat[(size_t)(ny * WDIM + nx) * 64 + ch];
      } else if (k == 576) v = (_Float16)s_r0[c][mm];
      else if (k == 577) v = (_Float16)s_r1[c][mm];
      else if (k == 578) v = (_Float16)s_rc0[mm];
      else if (k == 579) v = (_Float16)s_rc1[mm];
      xt[mm][k] = v;
    }
    __syncthreads();
    gemm_lds(&xt[0][0], 608, w0, 608, b0, 256, 19, 4, &hb[0][0], 256, true, lane);
    __syncthreads();
    gemm_lds(&hb[0][0], 256, w1, 256, b1, 256, 8, 4, &hc[0][0], 256, true, lane);
    __syncthreads();
    gemm_lds(&hc[0][0], 256, w2, 256, b2, 256, 8, 4, &hb[0][0], 256, true, lane);
    __syncthreads();
    gemm_lds(&hb[0][0], 256, w3, 256, b3, 256, 8, 4, &hc[0][0], 256, true, lane);
    __syncthreads();
    // final layer (N=3 padded to 16) + weighted accumulation of preds
    {
      const int m = lane & 15, hi = lane >> 4;
      float bv = (m < 3) ? b4[m] : 0.f;
      v8f acc = {bv, bv, bv, bv, bv, bv, bv, bv};
#pragma clang loop unroll_count(2)
      for (int ks = 0; ks < 8; ++ks) {
        v16h a = load_a_frag(&hc[m][0], ks, lane);
        v16h b = load_b_frag(w4 + (size_t)m * 256, ks, lane);
        acc = wmma_step(a, b, acc);
      }
      if (m < 3) {
#pragma unroll
        for (int v = 0; v < 8; ++v) {
          int row = v + hi * 8;
          accP[row][m] += s_w[c][row] * acc[v];
        }
      }
    }
    // weighted accumulation of inter (hc)
    for (int e = lane; e < 16 * 256; e += 32) {
      int mm = e >> 8; int j = e & 255;
      accI[mm][j] += s_w[c][mm] * (float)hc[mm][j];
    }
    __syncthreads();
  }

  for (int e = lane; e < 16 * 3; e += 32) {
    int mm = e / 3, j = e - mm * 3;
    out_ldr[(size_t)(qbase + mm) * 3 + j] = accP[mm][j];
  }
  for (int e = lane; e < 16 * 256; e += 32) {
    int mm = e >> 8; int j = e & 255;
    f4h[(size_t)(qbase + mm) * 256 + j] = (_Float16)accI[mm][j];
  }
}

// ---------------- HDR MLP ----------------
__global__ __launch_bounds__(32, 1) void hdr_k(
    const _Float16* __restrict__ f4h,    // [Q][256]
    const float* __restrict__ gcoord,    // [Q][2]
    const _Float16* __restrict__ w0, const float* __restrict__ b0,
    const _Float16* __restrict__ w1, const float* __restrict__ b1,
    const _Float16* __restrict__ w2, const float* __restrict__ b2,
    float* __restrict__ out_hdr) {       // [Q][3]
  __shared__ alignas(32) _Float16 xt[16][288];
  __shared__ alignas(32) _Float16 hb[16][256];
  __shared__ alignas(32) _Float16 hc[16][256];
  const int lane = threadIdx.x;
  const int qbase = blockIdx.x * 16;

  for (int e = lane; e < 16 * 288; e += 32) {
    int mm = e / 288; int k = e - mm * 288;
    int q = qbase + mm;
    _Float16 v = (_Float16)0;
    if (k < 256) v = f4h[(size_t)q * 256 + k];
    else if (k == 256) v = (_Float16)gcoord[q * 2 + 0];
    else if (k == 257) v = (_Float16)gcoord[q * 2 + 1];
    xt[mm][k] = v;
  }
  __syncthreads();
  gemm_lds(&xt[0][0], 288, w0, 288, b0, 256, 9, 4, &hb[0][0], 256, true, lane);
  __syncthreads();
  gemm_lds(&hb[0][0], 256, w1, 256, b1, 256, 8, 4, &hc[0][0], 256, true, lane);
  __syncthreads();
  {
    const int m = lane & 15, hi = lane >> 4;
    float bv = (m < 3) ? b2[m] : 0.f;
    v8f acc = {bv, bv, bv, bv, bv, bv, bv, bv};
#pragma clang loop unroll_count(2)
    for (int ks = 0; ks < 8; ++ks) {
      v16h a = load_a_frag(&hc[m][0], ks, lane);
      v16h b = load_b_frag(w2 + (size_t)m * 256, ks, lane);
      acc = wmma_step(a, b, acc);
    }
    if (m < 3) {
#pragma unroll
      for (int v = 0; v < 8; ++v) {
        int row = v + hi * 8;
        out_hdr[(size_t)(qbase + row) * 3 + m] = acc[v];
      }
    }
  }
}

// ---------------- weight / input prep ----------------
__global__ void prep_input_k(const float* __restrict__ lr, _Float16* __restrict__ xin) {
  int idx = blockIdx.x * blockDim.x + threadIdx.x;
  if (idx >= HWPIX * 3) return;
  int p = idx / 3, c = idx - p * 3;
  const float mean[3] = {0.4488f, 0.4371f, 0.4040f};
  xin[idx] = (_Float16)(lr[(size_t)c * HWPIX + p] - mean[c]);
}

// OIHW fp32 -> Wt[co][k] f16, k = (ky*3+kx)*CI + ci, padded to KPAD
__global__ void prep_convw(const float* __restrict__ src, _Float16* __restrict__ dst,
                           int CO, int CI, int KPAD) {
  int idx = blockIdx.x * blockDim.x + threadIdx.x;
  if (idx >= CO * KPAD) return;
  int co = idx / KPAD, k = idx - co * KPAD;
  float v = 0.f;
  if (k < 9 * CI) {
    int tap = k / CI, ci = k - tap * CI;
    int ky = tap / 3, kx = tap - ky * 3;
    v = src[(((size_t)co * CI + ci) * 3 + ky) * 3 + kx];
  }
  dst[idx] = (_Float16)v;
}

// [KIN][NOUT] fp32 -> Wt[n][k] f16, rows padded to NROWS, K padded to KPAD
__global__ void prep_mlpw(const float* __restrict__ src, _Float16* __restrict__ dst,
                          int KIN, int NOUT, int NROWS, int KPAD) {
  int idx = blockIdx.x * blockDim.x + threadIdx.x;
  if (idx >= NROWS * KPAD) return;
  int n = idx / KPAD, k = idx - n * KPAD;
  float v = 0.f;
  if (k < KIN && n < NOUT) v = src[(size_t)k * NOUT + n];
  dst[idx] = (_Float16)v;
}

// ---------------- host-side orchestration ----------------
extern "C" void kernel_launch(void* const* d_in, const int* in_sizes, int n_in,
                              void* d_out, int out_size, void* d_ws, size_t ws_size,
                              hipStream_t stream) {
  const float* lr_ldr = (const float*)d_in[0];
  const float* local_coord = (const float*)d_in[1];
  const float* cell = (const float*)d_in[2];
  const float* gcoord = (const float*)d_in[3];
  const float* head_w = (const float*)d_in[4];
  const float* head_b = (const float*)d_in[5];
  const float* rb_w1 = (const float*)d_in[6];
  const float* rb_b1 = (const float*)d_in[7];
  const float* rb_w2 = (const float*)d_in[8];
  const float* rb_b2 = (const float*)d_in[9];
  const float* tail_w = (const float*)d_in[10];
  const float* tail_b = (const float*)d_in[11];
  const float* srw[5] = {(const float*)d_in[12], (const float*)d_in[14], (const float*)d_in[16],
                         (const float*)d_in[18], (const float*)d_in[20]};
  const float* srb[5] = {(const float*)d_in[13], (const float*)d_in[15], (const float*)d_in[17],
                         (const float*)d_in[19], (const float*)d_in[21]};
  const float* hdw[3] = {(const float*)d_in[22], (const float*)d_in[24], (const float*)d_in[26]};
  const float* hdb[3] = {(const float*)d_in[23], (const float*)d_in[25], (const float*)d_in[27]};

  char* ws = (char*)d_ws;
  size_t off = 0;
  auto alloc = [&](size_t bytes) -> void* {
    off = (off + 255) & ~(size_t)255;
    void* p = ws + off;
    off += bytes;
    return p;
  };

  _Float16* xin    = (_Float16*)alloc((size_t)HWPIX * 3 * 2);
  _Float16* h0     = (_Float16*)alloc((size_t)HWPIX * 64 * 2);
  _Float16* res    = (_Float16*)alloc((size_t)HWPIX * 64 * 2);
  _Float16* tbuf   = (_Float16*)alloc((size_t)HWPIX * 64 * 2);
  _Float16* feat   = (_Float16*)alloc((size_t)HWPIX * 64 * 2);
  _Float16* wt_head = (_Float16*)alloc((size_t)64 * 32 * 2);
  _Float16* wt_rb1  = (_Float16*)alloc((size_t)16 * 64 * 576 * 2);
  _Float16* wt_rb2  = (_Float16*)alloc((size_t)16 * 64 * 576 * 2);
  _Float16* wt_tail = (_Float16*)alloc((size_t)64 * 576 * 2);
  _Float16* wt_sr0  = (_Float16*)alloc((size_t)256 * 608 * 2);
  _Float16* wt_sr1  = (_Float16*)alloc((size_t)256 * 256 * 2);
  _Float16* wt_sr2  = (_Float16*)alloc((size_t)256 * 256 * 2);
  _Float16* wt_sr3  = (_Float16*)alloc((size_t)256 * 256 * 2);
  _Float16* wt_sr4  = (_Float16*)alloc((size_t)16 * 256 * 2);
  _Float16* wt_hd0  = (_Float16*)alloc((size_t)256 * 288 * 2);
  _Float16* wt_hd1  = (_Float16*)alloc((size_t)256 * 256 * 2);
  _Float16* wt_hd2  = (_Float16*)alloc((size_t)16 * 256 * 2);
  _Float16* f4h     = (_Float16*)alloc((size_t)QDIM * 256 * 2);
  if (off > ws_size) return;  // scratch too small; avoid OOB

  const int T = 256;
  prep_input_k<<<(HWPIX * 3 + T - 1) / T, T, 0, stream>>>(lr_ldr, xin);
  prep_convw<<<(64 * 32 + T - 1) / T, T, 0, stream>>>(head_w, wt_head, 64, 3, 32);
  for (int i = 0; i < 16; ++i) {
    prep_convw<<<(64 * 576 + T - 1) / T, T, 0, stream>>>(
        rb_w1 + (size_t)i * 64 * 64 * 9, wt_rb1 + (size_t)i * 64 * 576, 64, 64, 576);
    prep_convw<<<(64 * 576 + T - 1) / T, T, 0, stream>>>(
        rb_w2 + (size_t)i * 64 * 64 * 9, wt_rb2 + (size_t)i * 64 * 576, 64, 64, 576);
  }
  prep_convw<<<(64 * 576 + T - 1) / T, T, 0, stream>>>(tail_w, wt_tail, 64, 64, 576);
  prep_mlpw<<<(256 * 608 + T - 1) / T, T, 0, stream>>>(srw[0], wt_sr0, 580, 256, 256, 608);
  prep_mlpw<<<(256 * 256 + T - 1) / T, T, 0, stream>>>(srw[1], wt_sr1, 256, 256, 256, 256);
  prep_mlpw<<<(256 * 256 + T - 1) / T, T, 0, stream>>>(srw[2], wt_sr2, 256, 256, 256, 256);
  prep_mlpw<<<(256 * 256 + T - 1) / T, T, 0, stream>>>(srw[3], wt_sr3, 256, 256, 256, 256);
  prep_mlpw<<<(16 * 256 + T - 1) / T, T, 0, stream>>>(srw[4], wt_sr4, 256, 3, 16, 256);
  prep_mlpw<<<(256 * 288 + T - 1) / T, T, 0, stream>>>(hdw[0], wt_hd0, 258, 256, 256, 288);
  prep_mlpw<<<(256 * 256 + T - 1) / T, T, 0, stream>>>(hdw[1], wt_hd1, 256, 256, 256, 256);
  prep_mlpw<<<(16 * 256 + T - 1) / T, T, 0, stream>>>(hdw[2], wt_hd2, 256, 3, 16, 256);

  // EDSR encoder
  const int CB = HWPIX / 32;  // 32 pixels per block (2 waves x 16)
  conv3x3_k<3, 0><<<CB, 64, 0, stream>>>(xin, wt_head, head_b, h0, nullptr);
  hipMemcpyAsync(res, h0, (size_t)HWPIX * 64 * 2, hipMemcpyDeviceToDevice, stream);
  for (int i = 0; i < 16; ++i) {
    conv3x3_k<64, 1><<<CB, 64, 0, stream>>>(res, wt_rb1 + (size_t)i * 64 * 576,
                                            rb_b1 + (size_t)i * 64, tbuf, nullptr);
    conv3x3_k<64, 2><<<CB, 64, 0, stream>>>(tbuf, wt_rb2 + (size_t)i * 64 * 576,
                                            rb_b2 + (size_t)i * 64, res, nullptr);
  }
  conv3x3_k<64, 3><<<CB, 64, 0, stream>>>(res, wt_tail, tail_b, feat, h0);

  // LIIF query + HDR heads
  float* out_ldr = (float*)d_out;
  float* out_hdr = (float*)d_out + (size_t)QDIM * 3;
  sr_query_k<<<QDIM / 16, 32, 0, stream>>>(feat, local_coord, cell,
                                           wt_sr0, srb[0], wt_sr1, srb[1], wt_sr2, srb[2],
                                           wt_sr3, srb[3], wt_sr4, srb[4],
                                           out_ldr, f4h);
  hdr_k<<<QDIM / 16, 32, 0, stream>>>(f4h, gcoord, wt_hd0, hdb[0], wt_hd1, hdb[1],
                                      wt_hd2, hdb[2], out_hdr);
  (void)in_sizes; (void)n_in; (void)out_size;
}
